// MultiHeadSGATLayer_10093173145796
// MI455X (gfx1250) — matrix operand order repaired and checked by reading, compiled, not run on hardware
//
#include <hip/hip_runtime.h>

// ---------------------------------------------------------------------------
// MultiHeadSGATLayer for MI455X (gfx1250, wave32, WMMA).
//
//   prep_w   : W fp32 [4,256,64] -> bf16 B-fragments in WMMA lane order (ws)
//   gat_gemm : z = h @ W_h via v_wmma_f32_16x16x32_bf16 (f32 accum).
//              A tile double-buffered through LDS (1 barrier/K-step);
//              B fragments double-buffered in registers (prefetch K-step t+1
//              before the step-t WMMAs, so load latency hides in-wave).
//              Fused es/ed = z . a_src / a_dst epilogue.
//   gat_agg  : 4 nodes per block; per-node softmax over 16 edges (16-lane
//              shfl groups); weighted gather-sum of z[src] with b128 gathers
//              (z = 51.2 MB lives in the 192 MB L2).
// ---------------------------------------------------------------------------

#define N_NODES 50000
#define DEG     16
#define IN_DIM  256
#define OUT_DIM 64
#define HEADS   4
#define SLOPE   0.01f

typedef __attribute__((ext_vector_type(16))) __bf16        v16bf;
typedef __attribute__((ext_vector_type(8)))  float         v8f;
typedef __attribute__((ext_vector_type(8)))  unsigned int  v8u;

__device__ __forceinline__ unsigned int f2bf1(float f) {
    // round-to-nearest-even fp32 -> bf16
    unsigned int u = __float_as_uint(f);
    return (u + 0x7FFFu + ((u >> 16) & 1u)) >> 16;
}
__device__ __forceinline__ unsigned int packbf(float lo, float hi) {
    return f2bf1(lo) | (f2bf1(hi) << 16);
}

// --------------------------------------------------------------------------
// Pre-swizzle W into bf16 B-matrix fragments.
// Fragment dword index: ((((h*8 + t)*4 + c)*32 + lane)*8 + v)
//   lane 0-15  hold K = t*32 + [0..15],  lane 16-31 hold K = t*32 + [16..31]
//   dword v packs B[kb+2v][n], B[kb+2v+1][n];  n = c*16 + (lane&15)
// --------------------------------------------------------------------------
__global__ __launch_bounds__(256) void prep_w(const float* __restrict__ W,
                                              unsigned int* __restrict__ Wb) {
    const int idx = blockIdx.x * 256 + threadIdx.x;     // 0..32767
    const int v = idx & 7;
    const int L = (idx >> 3) & 31;
    const int c = (idx >> 8) & 3;
    const int t = (idx >> 10) & 7;
    const int hh = idx >> 13;
    const int k = t * 32 + ((L >> 4) << 4) + 2 * v;
    const int n = c * 16 + (L & 15);
    const float* Wh = W + (size_t)hh * IN_DIM * OUT_DIM;
    const float w0 = Wh[(size_t)k * OUT_DIM + n];
    const float w1 = Wh[(size_t)(k + 1) * OUT_DIM + n];
    Wb[idx] = packbf(w0, w1);
}

// --------------------------------------------------------------------------
// GEMM: block = 128 threads = 4 waves; wave w == head w, rows
// [16*blockIdx .. +16), 64 head-columns (4 WMMA C tiles), K = 256 in 8 steps
// of 32. A tile (shared by all 4 waves) double-buffered in LDS; B fragments
// double-buffered in VGPRs.
// --------------------------------------------------------------------------
__global__ __launch_bounds__(128) void gat_gemm(
    const float* __restrict__ hfeat, const unsigned int* __restrict__ Wb,
    const float* __restrict__ a_src, const float* __restrict__ a_dst,
    float* __restrict__ z, float* __restrict__ es, float* __restrict__ ed) {

    __shared__ unsigned int sA[2][16 * 16];  // 2 x (16 rows x 32 bf16)

    const int tid  = threadIdx.x;
    const int lane = tid & 31;
    const int head = tid >> 5;             // wave id == head
    const int m0   = blockIdx.x * 16;
    const int half = lane >> 4;            // 0: lanes 0-15, 1: lanes 16-31
    const int ln   = lane & 15;

    v8f acc[4] = {};

    const int arow = tid >> 3;             // 0..15 (A-tile staging row)
    const int acol = tid & 7;              // 0..7  (x4 floats)
    const float* hrow = hfeat + (size_t)(m0 + arow) * IN_DIM + acol * 4;
    const int spos = arow * 16 + acol * 2;

    // prologue: stage A tile t=0
    {
        const float4 hv = *(const float4*)hrow;
        sA[0][spos + 0] = packbf(hv.x, hv.y);
        sA[0][spos + 1] = packbf(hv.z, hv.w);
    }

    // prologue: preload B fragments for t=0 (per-lane 32B per column tile)
    const unsigned int* wbBase = Wb + ((size_t)(head * 8 * 4) * 32 + lane) * 8;
    v8u bbuf[2][4];
#pragma unroll
    for (int c = 0; c < 4; ++c) bbuf[0][c] = *(const v8u*)(wbBase + c * 256);

    const int base0 = ln * 16 + half * 8;  // A fragment dword base in tile

#pragma unroll
    for (int t = 0; t < 8; ++t) {
        __syncthreads();                   // sA[t&1] ready for reads

        // issue next A-tile global load early (overlaps with WMMAs below)
        float4 hv;
        if (t < 7) hv = *(const float4*)(hrow + (t + 1) * 32);

        // prefetch next K-step's B fragments into the alternate register buf
        if (t < 7) {
#pragma unroll
            for (int c = 0; c < 4; ++c)
                bbuf[(t + 1) & 1][c] =
                    *(const v8u*)(wbBase + (t + 1) * 1024 + c * 256);
        }

        // A fragment (contiguous K order, matches B): row = ln,
        // lanes 0-15: K 0-15 ; lanes 16-31: K 16-31
        const unsigned int* sb = sA[t & 1];
        const uint4 a0 = *(const uint4*)&sb[base0];
        const uint4 a1 = *(const uint4*)&sb[base0 + 4];
        const v8u au = {a0.x, a0.y, a0.z, a0.w, a1.x, a1.y, a1.z, a1.w};
        const v16bf afrag = __builtin_bit_cast(v16bf, au);

#pragma unroll
        for (int c = 0; c < 4; ++c) {
            const v16bf bfrag = __builtin_bit_cast(v16bf, bbuf[t & 1][c]);
            acc[c] = __builtin_amdgcn_wmma_f32_16x16x32_bf16(
                false, afrag, false, bfrag, (short)0, acc[c], false, false);
        }

        // stage t+1 into the other LDS buffer (nobody reads it this iter)
        if (t < 7) {
            sA[(t + 1) & 1][spos + 0] = packbf(hv.x, hv.y);
            sA[(t + 1) & 1][spos + 1] = packbf(hv.z, hv.w);
        }
    }

    // Epilogue: C layout — VGPR j: lanes 0-15 = (M=j, N=ln), lanes 16-31 = (M=j+8)
    float se[8] = {}, sd[8] = {};
#pragma unroll
    for (int c = 0; c < 4; ++c) {
        const float as = a_src[head * OUT_DIM + c * 16 + ln];
        const float ad = a_dst[head * OUT_DIM + c * 16 + ln];
#pragma unroll
        for (int j = 0; j < 8; ++j) {
            const int row = m0 + j + half * 8;
            z[(size_t)row * (HEADS * OUT_DIM) + head * OUT_DIM + c * 16 + ln] = acc[c][j];
            se[j] += acc[c][j] * as;
            sd[j] += acc[c][j] * ad;
        }
    }
    // reduce over the 16 N-lanes of each half-wave
#pragma unroll
    for (int j = 0; j < 8; ++j) {
#pragma unroll
        for (int off = 8; off >= 1; off >>= 1) {
            se[j] += __shfl_xor(se[j], off, 32);
            sd[j] += __shfl_xor(sd[j], off, 32);
        }
    }
    if (ln == 0) {
#pragma unroll
        for (int j = 0; j < 8; ++j) {
            const int row = m0 + j + half * 8;
            es[row * HEADS + head] = se[j];
            ed[row * HEADS + head] = sd[j];
        }
    }
}

// --------------------------------------------------------------------------
// Aggregation: 4 nodes per 256-thread block.
//   Phase 1: tid = nl*64 + hh*16 + j  -> softmax over the 16 consecutive
//            incoming edges of node (blockIdx*4+nl), head hh (16-lane shfl).
//   Phase 2: thread owns 4 channels (float4); 16 b128 gathers from z.
// --------------------------------------------------------------------------
__global__ __launch_bounds__(256) void gat_agg(
    const float* __restrict__ z, const float* __restrict__ es,
    const float* __restrict__ ed, const int* __restrict__ src,
    float* __restrict__ out) {

    const int tid = threadIdx.x;
    const int nl  = tid >> 6;                      // node within block: 0..3
    const int n   = blockIdx.x * 4 + nl;

    __shared__ int   s_src[4 * DEG];               // 64 edges
    __shared__ float s_alpha[4][HEADS * DEG];

    if (tid < 4 * DEG) s_src[tid] = src[blockIdx.x * (4 * DEG) + tid];
    __syncthreads();

    // softmax: 4 nodes x 4 heads x 16 edges == 256 threads
    {
        const int hh = (tid >> 4) & 3;
        const int j  = tid & 15;
        float e = es[s_src[nl * DEG + j] * HEADS + hh] + ed[n * HEADS + hh];
        e = (e >= 0.f) ? e : SLOPE * e;            // leaky relu
        float m = e;
#pragma unroll
        for (int off = 8; off >= 1; off >>= 1) m = fmaxf(m, __shfl_xor(m, off, 32));
        const float ex = __expf(e - m);
        float s = ex;
#pragma unroll
        for (int off = 8; off >= 1; off >>= 1) s += __shfl_xor(s, off, 32);
        s_alpha[nl][hh * DEG + j] = ex / s;
    }
    __syncthreads();

    // weighted gather-sum: 4 channels per thread
    const int cbase = (tid & 63) * 4;              // 0,4,...,252
    const int hh    = cbase >> 6;                  // head of these channels
    int   sj[DEG];
    float al[DEG];
#pragma unroll
    for (int j = 0; j < DEG; ++j) {
        sj[j] = s_src[nl * DEG + j];
        al[j] = s_alpha[nl][hh * DEG + j];
    }

    float4 acc = {0.f, 0.f, 0.f, 0.f};
#pragma unroll
    for (int j = 0; j < DEG; ++j) {
        const float4 zv = *(const float4*)(z + (size_t)sj[j] * (HEADS * OUT_DIM) + cbase);
        acc.x += al[j] * zv.x;
        acc.y += al[j] * zv.y;
        acc.z += al[j] * zv.z;
        acc.w += al[j] * zv.w;
    }
    *(float4*)(out + (size_t)n * (HEADS * OUT_DIM) + cbase) = acc;
}

// --------------------------------------------------------------------------
extern "C" void kernel_launch(void* const* d_in, const int* in_sizes, int n_in,
                              void* d_out, int out_size, void* d_ws, size_t ws_size,
                              hipStream_t stream) {
    const float* h     = (const float*)d_in[0];   // [N,256]
    const float* W     = (const float*)d_in[1];   // [4,256,64]
    const float* a_src = (const float*)d_in[2];   // [4,64]
    const float* a_dst = (const float*)d_in[3];   // [4,64]
    const int*   src   = (const int*)d_in[4];     // [E]
    // d_in[5] = dst (known structure: repeat(arange(N), 16)) — not needed

    float* out = (float*)d_out;                   // [N, 256]

    // workspace layout
    float* z  = (float*)d_ws;                             // N*256 f32 (51.2 MB)
    float* es = z + (size_t)N_NODES * HEADS * OUT_DIM;    // N*4 f32
    float* ed = es + (size_t)N_NODES * HEADS;             // N*4 f32
    unsigned int* Wb = (unsigned int*)(ed + (size_t)N_NODES * HEADS); // 128 KB

    prep_w  <<<128, 256, 0, stream>>>(W, Wb);
    gat_gemm<<<N_NODES / 16, 128, 0, stream>>>(h, Wb, a_src, a_dst, z, es, ed);
    gat_agg <<<N_NODES / 4, 256, 0, stream>>>(z, es, ed, src, out);
}
